// BiLSTM_CRF_16784732193091
// MI455X (gfx1250) — matrix operand order repaired and checked by reading, not compiled, not run on hardware
//
#include <hip/hip_runtime.h>
#include <hip/hip_bf16.h>
#include <stdint.h>

#define B_    64
#define S_    256
#define E_    256
#define HALF_ 256
#define HDIM_ 512
#define G4_   (4 * HALF_)   // 1024 gate width per direction
#define TAGS_ 32
#define NEG_  (-10000.0f)
#define BS_   (B_ * S_)     // 16384 rows

typedef __bf16 bf16;
typedef bf16  v8bf  __attribute__((ext_vector_type(8)));
typedef bf16  v16bf __attribute__((ext_vector_type(16)));
typedef float v8f   __attribute__((ext_vector_type(8)));

static __device__ __forceinline__ v16bf combine16(v8bf lo, v8bf hi) {
    v16bf r;
#pragma unroll
    for (int i = 0; i < 8; ++i) { r[i] = lo[i]; r[i + 8] = hi[i]; }
    return r;
}

static __device__ __forceinline__ float sigf(float x) {
    return 1.0f / (1.0f + __expf(-x));
}

// ---------------------------------------------------------------------------
// fp32 -> bf16 conversion (weights)
// ---------------------------------------------------------------------------
__global__ void f32_to_bf16_kernel(const float* __restrict__ src,
                                   bf16* __restrict__ dst, int n) {
    int i = blockIdx.x * blockDim.x + threadIdx.x;
    if (i < n) dst[i] = (bf16)src[i];
}

// ---------------------------------------------------------------------------
// Embedding gather -> bf16 x[B*S, E]
// ---------------------------------------------------------------------------
__global__ void gather_embed_kernel(const int* __restrict__ sent,
                                    const float* __restrict__ embed,
                                    bf16* __restrict__ x_bf) {
    int row = blockIdx.x;          // 0 .. B*S-1
    int e   = threadIdx.x;         // 0 .. 255
    int tok = sent[row];
    x_bf[(size_t)row * E_ + e] = (bf16)embed[(size_t)tok * E_ + e];
}

// ---------------------------------------------------------------------------
// Generic WMMA bf16 GEMM: C[M,N] = A[M,K] * B[N,K]^T + bias[N]
// One 16x16 tile per wave; K multiple of 32, M,N multiples of 16.
// ---------------------------------------------------------------------------
__global__ __launch_bounds__(256)
void wmma_gemm_bias_kernel(const bf16* __restrict__ A,
                           const bf16* __restrict__ Bw,
                           const float* __restrict__ bias,
                           float* __restrict__ C,
                           int M, int N, int K) {
    int lane = threadIdx.x & 31;
    int wave = threadIdx.x >> 5;
    int tile = blockIdx.x * (blockDim.x >> 5) + wave;
    int ntn  = N >> 4;
    int total = (M >> 4) * ntn;
    if (tile >= total) return;                     // wave-uniform guard
    int m0 = (tile / ntn) << 4;
    int n0 = (tile % ntn) << 4;

    int l15 = lane & 15;
    int hiA = (lane >> 4) ? 16 : 0;                // A: upper half-wave K offset
    int hiB = (lane >> 4) ? 16 : 0;                // B: upper half-wave K offset

    const bf16* arow = A + (size_t)(m0 + l15) * K;
    const bf16* brow = Bw + (size_t)(n0 + l15) * K;

    v8f acc = {};
    for (int kc = 0; kc < K; kc += 32) {
        v8bf alo = *(const v8bf*)(arow + kc + hiA);
        v8bf ahi = *(const v8bf*)(arow + kc + hiA + 8);
        v16bf av = combine16(alo, ahi);
        v16bf bv = *(const v16bf*)(brow + kc + hiB);
        acc = __builtin_amdgcn_wmma_f32_16x16x32_bf16(
                  false, av, false, bv, (short)0, acc, false, false);
    }

    int ncol = n0 + l15;
    int mhi  = (lane >> 4) << 3;
    float bb = bias ? bias[ncol] : 0.0f;
#pragma unroll
    for (int r = 0; r < 8; ++r)
        C[(size_t)(m0 + mhi + r) * N + ncol] = acc[r] + bb;
}

// ---------------------------------------------------------------------------
// Persistent bidirectional LSTM.
// grid = (4 batch tiles, 2 directions), block = 1024 threads (32 waves).
// Per step: gates[16,1024] = [x_t | h] (16x512 bf16) @ [Wih | Whh]^T via WMMA,
// then elementwise cell update; h kept in LDS (bf16), c in registers.
// ---------------------------------------------------------------------------
#define HSTR_ (HALF_ + 8)     // padded bf16 stride for h buffer
#define ZSTR_ (G4_ + 4)       // padded f32 stride for gate tile

__global__ __launch_bounds__(1024)
void bilstm_kernel(const bf16* __restrict__ x_bf,       // [B*S, E]
                   const bf16* __restrict__ Wihf,       // [4H, E]
                   const bf16* __restrict__ Whhf,       // [4H, H]
                   const float* __restrict__ bf_,       // [4H]
                   const bf16* __restrict__ Wihb,
                   const bf16* __restrict__ Whhb,
                   const float* __restrict__ bb_,
                   bf16* __restrict__ Hout) {           // [B*S, HDIM]
    __shared__ float zl[16 * ZSTR_];                    // 16 x 1024 gate tile
    __shared__ bf16  hbuf[16 * HSTR_];                  // 16 x 256 hidden (bf16)

    const int dir = blockIdx.y;                         // 0 = fwd, 1 = bwd
    const int b0  = blockIdx.x << 4;                    // 16 batch rows per block
    const bf16*  Wih  = dir ? Wihb : Wihf;
    const bf16*  Whh  = dir ? Whhb : Whhf;
    const float* bias = dir ? bb_  : bf_;
    const int hcol = dir * HALF_;

    const int tid  = threadIdx.x;
    const int lane = tid & 31;
    const int wave = tid >> 5;
    const int l15  = lane & 15;
    const int hiA  = (lane >> 4) ? 16 : 0;
    const int hiB  = (lane >> 4) ? 16 : 0;

    // zero h buffer; c lives in 4 registers per thread (16*256/1024)
    for (int i = tid; i < 16 * HSTR_; i += 1024) hbuf[i] = (bf16)0.0f;
    float c[4] = {0.f, 0.f, 0.f, 0.f};
    __syncthreads();

    for (int step = 0; step < S_; ++step) {
        const int tpos = dir ? (S_ - 1 - step) : step;

        // ---------------- GEMM phase: zl = [x_t | h] @ [Wih | Whh]^T ------
        const bf16* arow = x_bf + ((size_t)(b0 + l15) * S_ + tpos) * E_;
        const bf16* hrow = hbuf + l15 * HSTR_;
#pragma unroll
        for (int tt = 0; tt < 2; ++tt) {
            const int n0 = ((wave << 1) + tt) << 4;
            v8f acc = {};
            // input contribution, K = 0..255
            const bf16* brow0 = Wih + (size_t)(n0 + l15) * E_;
#pragma unroll
            for (int kc = 0; kc < E_; kc += 32) {
                v8bf alo = *(const v8bf*)(arow + kc + hiA);
                v8bf ahi = *(const v8bf*)(arow + kc + hiA + 8);
                v16bf av = combine16(alo, ahi);
                v16bf bv = *(const v16bf*)(brow0 + kc + hiB);
                acc = __builtin_amdgcn_wmma_f32_16x16x32_bf16(
                          false, av, false, bv, (short)0, acc, false, false);
            }
            // recurrent contribution, K = 0..255 from LDS h
            const bf16* brow1 = Whh + (size_t)(n0 + l15) * HALF_;
#pragma unroll
            for (int kc = 0; kc < HALF_; kc += 32) {
                v8bf alo = *(const v8bf*)(hrow + kc + hiA);
                v8bf ahi = *(const v8bf*)(hrow + kc + hiA + 8);
                v16bf av = combine16(alo, ahi);
                v16bf bv = *(const v16bf*)(brow1 + kc + hiB);
                acc = __builtin_amdgcn_wmma_f32_16x16x32_bf16(
                          false, av, false, bv, (short)0, acc, false, false);
            }
            // spill accumulator tile to LDS (C layout: m = r + 8*(lane>>4))
            const int ncol = n0 + l15;
            const int mhi  = (lane >> 4) << 3;
#pragma unroll
            for (int r = 0; r < 8; ++r)
                zl[(mhi + r) * ZSTR_ + ncol] = acc[r];
        }
        __syncthreads();

        // ---------------- elementwise LSTM cell ---------------------------
#pragma unroll
        for (int q = 0; q < 4; ++q) {
            const int p = tid + (q << 10);      // 0..4095
            const int m = p >> 8;               // batch row in tile
            const int j = p & 255;              // hidden index
            const float* z = zl + m * ZSTR_;
            float zi = z[j]             + bias[j];
            float zf = z[256 + j]       + bias[256 + j];
            float zg = z[512 + j]       + bias[512 + j];
            float zo = z[768 + j]       + bias[768 + j];
            float cc = sigf(zf) * c[q] + sigf(zi) * tanhf(zg);
            float hh = sigf(zo) * tanhf(cc);
            c[q] = cc;
            bf16 hb = (bf16)hh;
            hbuf[m * HSTR_ + j] = hb;
            const size_t row = (size_t)(b0 + m) * S_ + tpos;
            Hout[row * HDIM_ + hcol + j] = hb;
        }
        __syncthreads();
    }
}

// ---------------------------------------------------------------------------
// Viterbi decode: one wave per batch row, one lane per tag.
// ---------------------------------------------------------------------------
__global__ __launch_bounds__(32)
void viterbi_kernel(const float* __restrict__ feats,   // [B*S, TAGS]
                    const float* __restrict__ trans,   // [TAGS, TAGS]
                    uint8_t* __restrict__ bp,          // [S, B, TAGS]
                    float* __restrict__ out_paths,     // [B, S]
                    float* __restrict__ out_scores) {  // [B]
    const int b = blockIdx.x;
    const int j = threadIdx.x;                         // next tag
    __shared__ float v[TAGS_];
    __shared__ float term[TAGS_];

    v[j] = (j == 0) ? 0.0f : NEG_;
    __syncthreads();

    for (int t = 0; t < S_; ++t) {
        float best = -3.0e38f;
        int arg = 0;
#pragma unroll
        for (int i = 0; i < TAGS_; ++i) {
            float s = v[i] + trans[i * TAGS_ + j];
            if (s > best) { best = s; arg = i; }       // first-max tie-break
        }
        bp[((size_t)t * B_ + b) * TAGS_ + j] = (uint8_t)arg;
        __syncthreads();
        v[j] = best + feats[((size_t)b * S_ + t) * TAGS_ + j];
        __syncthreads();
    }

    term[j] = v[j] + trans[j];                         // trans[0, j]
    __syncthreads();

    if (j == 0) {
        float best = -3.0e38f;
        int last = 0;
#pragma unroll
        for (int i = 0; i < TAGS_; ++i)
            if (term[i] > best) { best = term[i]; last = i; }
        out_scores[b] = best;
        out_paths[(size_t)b * S_ + (S_ - 1)] = (float)last;
        for (int t = S_ - 1; t >= 1; --t) {
            last = bp[((size_t)t * B_ + b) * TAGS_ + last];
            out_paths[(size_t)b * S_ + (t - 1)] = (float)last;
        }
    }
}

// ---------------------------------------------------------------------------
// Host launch
// ---------------------------------------------------------------------------
extern "C" void kernel_launch(void* const* d_in, const int* in_sizes, int n_in,
                              void* d_out, int out_size, void* d_ws, size_t ws_size,
                              hipStream_t stream) {
    const int*   sentence = (const int*)d_in[0];
    const float* embed    = (const float*)d_in[1];
    const float* Wih_f    = (const float*)d_in[2];
    const float* Whh_f    = (const float*)d_in[3];
    const float* b_f      = (const float*)d_in[4];
    const float* Wih_b    = (const float*)d_in[5];
    const float* Whh_b    = (const float*)d_in[6];
    const float* b_b      = (const float*)d_in[7];
    const float* Wout     = (const float*)d_in[8];
    const float* bout     = (const float*)d_in[9];
    const float* trans    = (const float*)d_in[10];

    // workspace layout (all 256B aligned)
    char* ws = (char*)d_ws;
    size_t off = 0;
    bf16* x_bf    = (bf16*)(ws + off); off += (size_t)BS_ * E_ * 2;        // 8 MB
    bf16* Wihf_bf = (bf16*)(ws + off); off += (size_t)G4_ * E_ * 2;        // 512 KB
    bf16* Whhf_bf = (bf16*)(ws + off); off += (size_t)G4_ * HALF_ * 2;
    bf16* Wihb_bf = (bf16*)(ws + off); off += (size_t)G4_ * E_ * 2;
    bf16* Whhb_bf = (bf16*)(ws + off); off += (size_t)G4_ * HALF_ * 2;
    bf16* Wout_bf = (bf16*)(ws + off); off += (size_t)TAGS_ * HDIM_ * 2;
    bf16* Hout    = (bf16*)(ws + off); off += (size_t)BS_ * HDIM_ * 2;     // 16 MB
    float* feats  = (float*)(ws + off); off += (size_t)BS_ * TAGS_ * 4;    // 2 MB
    uint8_t* bp   = (uint8_t*)(ws + off); off += (size_t)S_ * B_ * TAGS_;  // 512 KB
    (void)ws_size; (void)in_sizes; (void)n_in; (void)out_size;

    // 1) weight conversion fp32 -> bf16
    auto cvt = [&](const float* s, bf16* d, int n) {
        f32_to_bf16_kernel<<<(n + 255) / 256, 256, 0, stream>>>(s, d, n);
    };
    cvt(Wih_f, Wihf_bf, G4_ * E_);
    cvt(Whh_f, Whhf_bf, G4_ * HALF_);
    cvt(Wih_b, Wihb_bf, G4_ * E_);
    cvt(Whh_b, Whhb_bf, G4_ * HALF_);
    cvt(Wout,  Wout_bf, TAGS_ * HDIM_);

    // 2) embedding gather
    gather_embed_kernel<<<BS_, 256, 0, stream>>>(sentence, embed, x_bf);

    // 3) persistent bidirectional LSTM (4 batch tiles x 2 directions)
    bilstm_kernel<<<dim3(4, 2), 1024, 0, stream>>>(
        x_bf, Wihf_bf, Whhf_bf, b_f, Wihb_bf, Whhb_bf, b_b, Hout);

    // 4) feats = H @ Wout^T + bout   (M=16384, N=32, K=512)
    {
        int tiles = (BS_ / 16) * (TAGS_ / 16);              // 2048
        int blocks = (tiles + 7) / 8;                       // 8 waves/block
        wmma_gemm_bias_kernel<<<blocks, 256, 0, stream>>>(
            Hout, Wout_bf, bout, feats, BS_, TAGS_, HDIM_);
    }

    // 5) Viterbi decode -> d_out = [paths (B*S), scores (B)]
    float* out_paths  = (float*)d_out;
    float* out_scores = out_paths + (size_t)B_ * S_;
    viterbi_kernel<<<B_, 32, 0, stream>>>(feats, trans, bp, out_paths, out_scores);
}
